// GrailV1Model_28484223107671
// MI455X (gfx1250) — matrix operand order repaired and checked by reading, not compiled, not run on hardware
//
#include <hip/hip_runtime.h>
#include <hip/hip_bf16.h>
#include <stdint.h>

// ---------------------------------------------------------------------------
// GrailV1 GNN forward for MI455X (gfx1250, wave32, WMMA).
// All heavy GEMMs (N=128) go through v_wmma_f32_16x16x32_f16 with the concat /
// gather (x[row], x[col], edge_attr, u[edge_batch]) fused into the A-tile
// loader so the E x 512 concat matrices are never materialized in HBM.
// LDS traffic is fully 128-bit:
//   - staging: 8-half (16B) vectorized runs with a scalar fallback only at
//     segment/zero-pad boundaries (worknode K=300, workrelation K=100)
//   - A fragment: two 16B chunks per lane; B fragment: one 32B run per lane
//     (weights staged transposed, pre-converted to fp16 column-major).
// Large intermediates live in fp16 (halves HBM traffic); fp32 accumulate.
// ---------------------------------------------------------------------------

typedef __attribute__((ext_vector_type(16))) _Float16 v16h;
typedef __attribute__((ext_vector_type(8)))  _Float16 v8h;
typedef __attribute__((ext_vector_type(8)))  float    v8f;
typedef __attribute__((ext_vector_type(4)))  float    v4f;

#define BSc   128
#define NPGc  200
#define Nc    (BSc*NPGc)      // 25600 nodes
#define EPGc  3200
#define Ec    (BSc*EPGc)      // 409600 edges
#define EMBW  1280            // pooled_edge(128)+pooled_node(128)+qa(1024)

__device__ __forceinline__ float gelu_f(float x){
  const float c = 0.7978845608028654f;               // sqrt(2/pi)
  return 0.5f*x*(1.0f + tanhf(c*(x + 0.044715f*x*x*x)));
}

// A-matrix built from up to 4 concatenated segments (gathered rows).
struct Seg {
  const void* src;      // f32 or f16 base pointer
  const int*  gather;   // nullptr = identity row index
  int         ld;       // row stride in elements
  int         cols;     // columns this segment contributes
  int         type;     // 0 = f32 source, 1 = f16 source
};
struct SegList { Seg s[4]; int n; };

// ---------------------------------------------------------------------------
// WMMA GEMM: out[M,128] = act(concat(segs)[M,Kp] @ W[Kp,128] + bias)
// Wp is the fp16 weight stored COLUMN-major: Wp[c*Kp + k].
// block = 256 threads (8 waves), tile 128(M) x 128(N), K-step 32.
// Wave w computes rows [16w,16w+16) x all 128 cols -> 8 wmma accumulators.
// ---------------------------------------------------------------------------
__global__ __launch_bounds__(256)
void k_gemm_wmma(SegList segs, const _Float16* __restrict__ Wp,
                 const float* __restrict__ bias, int M, int Kp,
                 void* __restrict__ out, int out_f16, int act)
{
  __shared__ _Float16 As[128*32];   // A tile, row-major [row][k]
  __shared__ _Float16 Bt[128*32];   // B tile, TRANSPOSED [col][k]
  const int tid  = threadIdx.x;
  const int wv   = tid >> 5;
  const int lane = tid & 31;
  const int m0   = blockIdx.x * 128;
  const int mloc = lane & 15;
  const int half = lane >> 4;

  v8f acc[8] = {};

  for (int kk = 0; kk < Kp; kk += 32) {
    // ---- stage A tile (128 x 32): 512 runs of 8 halves, vectorized ----
    #pragma unroll
    for (int i = 0; i < 2; ++i) {
      int q  = tid + 256*i;           // run id
      int r  = q >> 2;                // tile row
      int c0 = (q & 3) << 3;          // run start column in k-chunk
      int off = kk + c0;
      // locate the segment containing the run start
      int loc = off, s = 0;
      const Seg* sg = nullptr;
      for (; s < segs.n; ++s) {
        if (loc < segs.s[s].cols) { sg = &segs.s[s]; break; }
        loc -= segs.s[s].cols;
      }
      _Float16* dst = &As[r*32 + c0];
      if (sg && loc + 8 <= sg->cols) {
        // fast path: whole run inside one segment -> 16B accesses
        int grow = m0 + r;
        if (sg->gather) grow = sg->gather[grow];
        long p = (long)grow * sg->ld + loc;
        if (sg->type) {
          *(v8h*)dst = *(const v8h*)((const _Float16*)sg->src + p);
        } else {
          const float* fp = (const float*)sg->src + p;
          v4f f0 = *(const v4f*)fp;
          v4f f1 = *(const v4f*)(fp + 4);
          v8h h;
          #pragma unroll
          for (int e = 0; e < 4; ++e) { h[e] = (_Float16)f0[e]; h[4+e] = (_Float16)f1[e]; }
          *(v8h*)dst = h;
        }
      } else {
        // slow path: run crosses a segment/zero-pad boundary
        #pragma unroll
        for (int e = 0; e < 8; ++e) {
          int o = off + e;
          float v = 0.f;
          for (int t2 = 0; t2 < segs.n; ++t2) {
            const Seg& g2 = segs.s[t2];
            if (o < g2.cols) {
              int grow = m0 + r;
              if (g2.gather) grow = g2.gather[grow];
              long pp = (long)grow * g2.ld + o;
              v = g2.type ? (float)((const _Float16*)g2.src)[pp]
                          : ((const float*)g2.src)[pp];
              break;
            }
            o -= g2.cols;
          }
          dst[e] = (_Float16)v;
        }
      }
    }
    // ---- stage B tile transposed (Bt[c*32+r] = W[kk+r, c]), 16B runs ----
    #pragma unroll
    for (int i = 0; i < 2; ++i) {
      int q    = tid + 256*i;
      int base = q << 3;              // element index in Bt (= c*32 + r)
      int c = base >> 5, r = base & 31;
      *(v8h*)&Bt[base] = *(const v8h*)&Wp[(long)c*Kp + kk + r];
    }
    __syncthreads();

    // A fragment: two contiguous 16B chunks per lane (b128 LDS loads).
    const int arow = (wv*16 + mloc)*32;
    v8h a0 = *(const v8h*)&As[arow + half*16];
    v8h a1 = *(const v8h*)&As[arow + half*16 + 8];
    v16h a = __builtin_shufflevector(a0, a1,
               0,1,2,3,4,5,6,7,8,9,10,11,12,13,14,15);
    #pragma unroll
    for (int n = 0; n < 8; ++n) {
      int colc = n*16 + mloc;
      // B fragment: one contiguous 32B run per lane (2x b128 LDS loads).
      v16h b = *(const v16h*)&Bt[colc*32 + half*16];
      acc[n] = __builtin_amdgcn_wmma_f32_16x16x32_f16(
                 false, a, false, b, (short)0, acc[n], false, false);
    }
    __syncthreads();
  }

  // ---- epilogue: bias + activation + store ----
  #pragma unroll
  for (int n = 0; n < 8; ++n) {
    int colc = n*16 + mloc;
    float bv = bias ? bias[colc] : 0.f;
    #pragma unroll
    for (int j = 0; j < 8; ++j) {
      long row = (long)m0 + wv*16 + half*8 + j;   // C layout: VGPR j -> M=j(+8)
      float v = acc[n][j] + bv;
      if (act) v = gelu_f(v);
      if (out_f16) ((_Float16*)out)[row*128 + colc] = (_Float16)v;
      else         ((float*)out)[row*128 + colc] = v;
    }
  }
}

// Small-N tail (Nout = 1 or 2): one thread per row.
__global__ void k_gemv(const _Float16* __restrict__ X, const float* __restrict__ W,
                       const float* __restrict__ b, int M, int K, int Nout,
                       float* __restrict__ out)
{
  int r = blockIdx.x*blockDim.x + threadIdx.x;
  if (r >= M) return;
  for (int n = 0; n < Nout; ++n) {
    float s = b[n];
    for (int k = 0; k < K; ++k) s += (float)X[(long)r*K + k] * W[k*Nout + n];
    out[(long)r*Nout + n] = s;
  }
}

// Pad/convert fp32 weight (K x 128 row-major) -> fp16 COLUMN-major [128][Kp],
// zero K-padding: dst[c*Kp + r] = W[r,c].
__global__ void k_pad_weight_t(const float* __restrict__ W, _Float16* __restrict__ dst,
                               int K, int Kp)
{
  int i = blockIdx.x*256 + threadIdx.x;
  if (i >= Kp*128) return;
  int c = i / Kp, r = i - c*Kp;
  dst[i] = (r < K) ? (_Float16)W[(long)r*128 + c] : (_Float16)0.f;
}

__global__ void k_egb(const int* __restrict__ row, const int* __restrict__ nb,
                      int* __restrict__ egb)
{
  int e = blockIdx.x*256 + threadIdx.x;
  if (e < Ec) egb[e] = nb[row[e]];
}

// Per-graph scatter-softmax over contiguous EPG edges + sigmoid (unnorm_w).
__global__ void k_edge_softmax(const float* __restrict__ wraw,
                               float* __restrict__ unw, float* __restrict__ nw)
{
  __shared__ float red[256];
  int b = blockIdx.x, t = threadIdx.x;
  int base = b*EPGc;
  float m = -3.0e38f;
  for (int e = t; e < EPGc; e += 256) m = fmaxf(m, wraw[base+e]);
  red[t] = m; __syncthreads();
  for (int s = 128; s > 0; s >>= 1) { if (t < s) red[t] = fmaxf(red[t], red[t+s]); __syncthreads(); }
  float gm = red[0]; __syncthreads();
  float sum = 0.f;
  for (int e = t; e < EPGc; e += 256) sum += expf(wraw[base+e] - gm);
  red[t] = sum; __syncthreads();
  for (int s = 128; s > 0; s >>= 1) { if (t < s) red[t] += red[t+s]; __syncthreads(); }
  float gs = red[0];
  for (int e = t; e < EPGc; e += 256) {
    float v = wraw[base+e];
    nw[base+e]  = expf(v - gm) / gs;
    unw[base+e] = 1.f / (1.f + expf(-v));
  }
}

// scatter-mean pass 1: atomic accumulate msg*norm_w into dest nodes + counts.
__global__ void k_scatter_add(const _Float16* __restrict__ msg,
                              const float* __restrict__ nw,
                              const int* __restrict__ col,
                              float* __restrict__ recv, float* __restrict__ cnt)
{
  long i = (long)blockIdx.x*256 + threadIdx.x;
  if (i >= (long)Ec*128) return;
  int e = (int)(i >> 7), d = (int)(i & 127);
  int c = col[e];
  atomicAdd(&recv[(long)c*128 + d], (float)msg[i]*nw[e]);
  if (d == 0) atomicAdd(&cnt[c], 1.f);
}

__global__ void k_div(float* __restrict__ recv, const float* __restrict__ cnt)
{
  int i = blockIdx.x*256 + threadIdx.x;
  if (i >= Nc*128) return;
  recv[i] /= fmaxf(cnt[i >> 7], 1.f);
}

// pooled_edge[b,:] = sum_e edge_attr[e,:]*norm_w[e] -> embeddings[:,0:128]
__global__ void k_pooled_edge(const _Float16* __restrict__ ea,
                              const float* __restrict__ nw, float* __restrict__ emb)
{
  int b = blockIdx.x, d = threadIdx.x;    // 128 threads
  long base = (long)b*EPGc;
  float s = 0.f;
  for (int e = 0; e < EPGc; ++e) s += (float)ea[(base+e)*128 + d]*nw[base+e];
  emb[(long)b*EMBW + d] = s;
}

// evidence = gelu(x) laid out [BS,NPG,128]; rows >= num_of_nodes zeroed.
__global__ void k_evidence(const _Float16* __restrict__ x, const int* __restrict__ non,
                           _Float16* __restrict__ ev)
{
  int i = blockIdx.x*256 + threadIdx.x;
  if (i >= Nc*128) return;
  int node = i >> 7;
  int b = node / NPGc, l = node - b*NPGc;
  float v = (l < non[b]) ? gelu_f((float)x[i]) : 0.f;
  ev[i] = (_Float16)v;
}

// 2-head attention pooling per graph -> embeddings[:,128:256]
__global__ void k_attention(const float* __restrict__ qs, const float* __restrict__ ks,
                            const float* __restrict__ vs, const int* __restrict__ non,
                            float* __restrict__ emb)
{
  __shared__ float sc[2][NPGc];
  int b = blockIdx.x, t = threadIdx.x;
  int nb = non[b];
  for (int p = t; p < 2*NPGc; p += 256) {
    int h = p / NPGc, l = p - h*NPGc;
    bool masked = (l >= nb) && !((nb <= 0) && (l == 0));
    float v;
    if (masked) v = -1e30f;
    else {
      float s = 0.f;
      const float* q = qs + (long)b*128 + h*64;
      const float* k = ks + ((long)b*NPGc + l)*128 + h*64;
      for (int d = 0; d < 64; ++d) s += q[d]*k[d];
      v = s * 0.125f;                 // 1/sqrt(64)
    }
    sc[h][l] = v;
  }
  __syncthreads();
  if (t < 2) {                        // softmax per head (L=200, cheap)
    float m = -3.0e38f;
    for (int l = 0; l < NPGc; ++l) m = fmaxf(m, sc[t][l]);
    float s = 0.f;
    for (int l = 0; l < NPGc; ++l) { float e = expf(sc[t][l]-m); sc[t][l] = e; s += e; }
    float inv = 1.f/s;
    for (int l = 0; l < NPGc; ++l) sc[t][l] *= inv;
  }
  __syncthreads();
  if (t < 128) {
    int h = t >> 6, d = t & 63;
    float s = 0.f;
    for (int l = 0; l < NPGc; ++l)
      s += sc[h][l]*vs[((long)b*NPGc + l)*128 + h*64 + d];
    emb[(long)b*EMBW + 128 + t] = s;
  }
}

__global__ void k_copy_qa(const float* __restrict__ qa, float* __restrict__ emb)
{
  int i = blockIdx.x*256 + threadIdx.x;
  if (i >= BSc*1024) return;
  int b = i >> 10, j = i & 1023;
  emb[(long)b*EMBW + 256 + j] = qa[i];
}

// ---------------------------------------------------------------------------
// host-side helpers
// ---------------------------------------------------------------------------
static inline char* ws_take(char*& p, size_t bytes) {
  char* r = (char*)(((uintptr_t)p + 255) & ~(uintptr_t)255);
  p = r + bytes;
  return r;
}
static inline Seg seg_f16(const _Float16* s, const int* g, int ld, int cols){ Seg x{s,g,ld,cols,1}; return x; }
static inline Seg seg_f32(const float*    s, const int* g, int ld, int cols){ Seg x{s,g,ld,cols,0}; return x; }
static inline SegList SL1(Seg a){ SegList L{}; L.s[0]=a; L.n=1; return L; }
static inline SegList SL2(Seg a, Seg b){ SegList L{}; L.s[0]=a; L.s[1]=b; L.n=2; return L; }
static inline SegList SL3(Seg a, Seg b, Seg c){ SegList L{}; L.s[0]=a; L.s[1]=b; L.s[2]=c; L.n=3; return L; }
static inline SegList SL4(Seg a, Seg b, Seg c, Seg d){ SegList L{}; L.s[0]=a; L.s[1]=b; L.s[2]=c; L.s[3]=d; L.n=4; return L; }

static inline void gemm(hipStream_t st, const SegList& segs, const _Float16* Wp,
                        const float* bias, int M, int Kp, void* out, int out_f16, int act)
{
  k_gemm_wmma<<<M/128, 256, 0, st>>>(segs, Wp, bias, M, Kp, out, out_f16, act);
}

extern "C" void kernel_launch(void* const* d_in, const int* in_sizes, int n_in,
                              void* d_out, int out_size, void* d_ws, size_t ws_size,
                              hipStream_t stream)
{
  (void)in_sizes; (void)n_in; (void)out_size; (void)ws_size;

  // ---- inputs (setup_inputs() dict insertion order, params flattened) ----
  const float* x_in  = (const float*)d_in[0];    // N x 300
  const int*   eidx  = (const int*)  d_in[1];    // 2 x E
  const float* ea_in = (const float*)d_in[2];    // E x 100
  const float* qa    = (const float*)d_in[3];    // BS x 1024
  const int*   nb    = (const int*)  d_in[4];    // N
  const int*   non   = (const int*)  d_in[5];    // BS
  const int* row = eidx;
  const int* col = eidx + Ec;

  const float *W_wn=(const float*)d_in[6],  *b_wn=(const float*)d_in[7];
  const float *W_wr=(const float*)d_in[8],  *b_wr=(const float*)d_in[9];
  const float *gW1=(const float*)d_in[10], *gb1=(const float*)d_in[11];
  const float *gW2=(const float*)d_in[12], *gb2=(const float*)d_in[13];
  const float *gW3=(const float*)d_in[14], *gb3=(const float*)d_in[15];
  const float *eW1=(const float*)d_in[16], *eb1=(const float*)d_in[17];
  const float *eW2=(const float*)d_in[18], *eb2=(const float*)d_in[19];
  const float *eW3=(const float*)d_in[20], *eb3=(const float*)d_in[21];
  const float *wW1=(const float*)d_in[22], *wb1=(const float*)d_in[23];
  const float *wW2=(const float*)d_in[24], *wb2=(const float*)d_in[25];
  const float *mW1=(const float*)d_in[26], *mb1=(const float*)d_in[27];
  const float *mW2=(const float*)d_in[28], *mb2=(const float*)d_in[29];
  const float *mW3=(const float*)d_in[30], *mb3=(const float*)d_in[31];
  const float *uW1=(const float*)d_in[32], *ub1=(const float*)d_in[33];
  const float *uW2=(const float*)d_in[34], *ub2=(const float*)d_in[35];
  const float *uW3=(const float*)d_in[36], *ub3=(const float*)d_in[37];
  const float *Wq =(const float*)d_in[38], *bq =(const float*)d_in[39];
  const float *Wk =(const float*)d_in[40], *bk =(const float*)d_in[41];
  const float *Wv =(const float*)d_in[42], *bv =(const float*)d_in[43];
  const float *hW1=(const float*)d_in[44], *hb1=(const float*)d_in[45];
  const float *hW2=(const float*)d_in[46], *hb2=(const float*)d_in[47];
  const float *hW3=(const float*)d_in[48], *hb3=(const float*)d_in[49];

  // ---- outputs: logits | unnorm_w | norm_w | embeddings (return order) ----
  float* out_logits = (float*)d_out;             // BS*2
  float* out_unw    = out_logits + BSc*2;        // E
  float* out_nw     = out_unw + Ec;              // E
  float* out_emb    = out_nw + Ec;               // BS*1280

  // ---- workspace carve ----
  char* wp = (char*)d_ws;
  _Float16* x_a  = (_Float16*)ws_take(wp, (size_t)Nc*128*2);
  _Float16* x_b  = (_Float16*)ws_take(wp, (size_t)Nc*128*2);
  _Float16* ea_a = (_Float16*)ws_take(wp, (size_t)Ec*128*2);
  _Float16* ea_b = (_Float16*)ws_take(wp, (size_t)Ec*128*2);
  _Float16* eh_a = (_Float16*)ws_take(wp, (size_t)Ec*128*2);
  _Float16* eh_b = (_Float16*)ws_take(wp, (size_t)Ec*128*2);
  _Float16* nh_a = (_Float16*)ws_take(wp, (size_t)Nc*128*2);
  _Float16* nh_b = (_Float16*)ws_take(wp, (size_t)Nc*128*2);
  _Float16* sh_a = (_Float16*)ws_take(wp, (size_t)128*128*2);
  _Float16* sh_b = (_Float16*)ws_take(wp, (size_t)128*128*2);
  float* u_g   = (float*)ws_take(wp, (size_t)BSc*128*4);
  float* recvb = (float*)ws_take(wp, (size_t)Nc*128*4);   // also reused as ks
  float* vsb   = (float*)ws_take(wp, (size_t)Nc*128*4);
  float* qsb   = (float*)ws_take(wp, (size_t)BSc*128*4);
  float* cntb  = (float*)ws_take(wp, (size_t)Nc*4);
  float* wraw  = (float*)ws_take(wp, (size_t)Ec*4);
  int*   egb   = (int*)  ws_take(wp, (size_t)Ec*4);
  _Float16* wpool = (_Float16*)ws_take(wp, (size_t)2*1024*1024);

  // ---- pad/convert all N=128 weights to fp16 column-major (zero K-pad) ----
  _Float16* cur = wpool;
  auto padw = [&](const float* W, int K, int Kp)->const _Float16* {
    _Float16* dst = cur; cur += (size_t)Kp*128;
    int tot = Kp*128;
    k_pad_weight_t<<<(tot+255)/256, 256, 0, stream>>>(W, dst, K, Kp);
    return dst;
  };
  const _Float16* p_wn = padw(W_wn, 300, 320);
  const _Float16* p_wr = padw(W_wr, 100, 128);
  const _Float16* p_g1 = padw(gW1, 1024, 1024);
  const _Float16* p_g2 = padw(gW2, 128, 128);
  const _Float16* p_g3 = padw(gW3, 128, 128);
  const _Float16* p_e1 = padw(eW1, 512, 512);
  const _Float16* p_e2 = padw(eW2, 128, 128);
  const _Float16* p_e3 = padw(eW3, 128, 128);
  const _Float16* p_w1 = padw(wW1, 256, 256);
  const _Float16* p_m1 = padw(mW1, 256, 256);
  const _Float16* p_m2 = padw(mW2, 128, 128);
  const _Float16* p_m3 = padw(mW3, 128, 128);
  const _Float16* p_u1 = padw(uW1, 384, 384);
  const _Float16* p_u2 = padw(uW2, 128, 128);
  const _Float16* p_u3 = padw(uW3, 128, 128);
  const _Float16* p_wq = padw(Wq, 1024, 1024);
  const _Float16* p_wk = padw(Wk, 128, 128);
  const _Float16* p_wv = padw(Wv, 128, 128);
  const _Float16* p_h1 = padw(hW1, 1280, 1280);
  const _Float16* p_h2 = padw(hW2, 128, 128);

  // edge -> graph id (edge_batch = node_batch[row])
  k_egb<<<(Ec+255)/256, 256, 0, stream>>>(row, nb, egb);

  // ---- input transforms ----
  gemm(stream, SL1(seg_f32(x_in,  nullptr, 300, 300)), p_wn, b_wn, Nc, 320, x_a, 1, 0);
  gemm(stream, SL1(seg_f32(ea_in, nullptr, 100, 100)), p_wr, b_wr, Ec, 128, ea_a, 1, 0);

  // ---- global MLP: u = MLP(qa) ----
  gemm(stream, SL1(seg_f32(qa, nullptr, 1024, 1024)), p_g1, gb1, BSc, 1024, sh_a, 1, 1);
  gemm(stream, SL1(seg_f16(sh_a, nullptr, 128, 128)), p_g2, gb2, BSc, 128, sh_b, 1, 1);
  gemm(stream, SL1(seg_f16(sh_b, nullptr, 128, 128)), p_g3, gb3, BSc, 128, u_g, 0, 0);

  // ---- 3 GNN layers ----
  _Float16* xc = x_a;  _Float16* xn = x_b;
  _Float16* ec = ea_a; _Float16* en = ea_b;
  for (int layer = 0; layer < 3; ++layer) {
    // weight_mlp on [edge_attr, u_e] -> wraw -> softmax/sigmoid
    gemm(stream, SL2(seg_f16(ec, nullptr, 128, 128), seg_f32(u_g, egb, 128, 128)),
         p_w1, wb1, Ec, 256, eh_a, 1, 1);
    k_gemv<<<(Ec+255)/256, 256, 0, stream>>>(eh_a, wW2, wb2, Ec, 128, 1, wraw);
    k_edge_softmax<<<BSc, 256, 0, stream>>>(wraw, out_unw, out_nw);

    // edge_mlp on [x[row], x[col], edge_attr, u_e] (gathers fused in GEMM)
    gemm(stream, SL4(seg_f16(xc, row, 128, 128), seg_f16(xc, col, 128, 128),
                     seg_f16(ec, nullptr, 128, 128), seg_f32(u_g, egb, 128, 128)),
         p_e1, eb1, Ec, 512, eh_a, 1, 1);
    gemm(stream, SL1(seg_f16(eh_a, nullptr, 128, 128)), p_e2, eb2, Ec, 128, eh_b, 1, 1);
    gemm(stream, SL1(seg_f16(eh_b, nullptr, 128, 128)), p_e3, eb3, Ec, 128, en, 1, 0);

    // message_mlp on [x[row], edge_attr_new]
    gemm(stream, SL2(seg_f16(xc, row, 128, 128), seg_f16(en, nullptr, 128, 128)),
         p_m1, mb1, Ec, 256, eh_a, 1, 1);
    gemm(stream, SL1(seg_f16(eh_a, nullptr, 128, 128)), p_m2, mb2, Ec, 128, eh_b, 1, 1);
    gemm(stream, SL1(seg_f16(eh_b, nullptr, 128, 128)), p_m3, mb3, Ec, 128, eh_a, 1, 0); // msg

    // scatter-mean of msg*norm_w into destination nodes
    (void)hipMemsetAsync(recvb, 0, (size_t)Nc*128*4, stream);
    (void)hipMemsetAsync(cntb,  0, (size_t)Nc*4, stream);
    k_scatter_add<<<((long)Ec*128)/256, 256, 0, stream>>>(eh_a, out_nw, col, recvb, cntb);
    k_div<<<(Nc*128+255)/256, 256, 0, stream>>>(recvb, cntb);

    // update_mlp on [x, recv, u[node_batch]]
    gemm(stream, SL3(seg_f16(xc, nullptr, 128, 128), seg_f32(recvb, nullptr, 128, 128),
                     seg_f32(u_g, nb, 128, 128)),
         p_u1, ub1, Nc, 384, nh_a, 1, 1);
    gemm(stream, SL1(seg_f16(nh_a, nullptr, 128, 128)), p_u2, ub2, Nc, 128, nh_b, 1, 1);
    gemm(stream, SL1(seg_f16(nh_b, nullptr, 128, 128)), p_u3, ub3, Nc, 128, xn, 1, 0);

    { _Float16* t = xc; xc = xn; xn = t; t = ec; ec = en; en = t; }
  }

  // ---- pooled edge vectors -> embeddings[:,0:128] ----
  k_pooled_edge<<<BSc, 128, 0, stream>>>(ec, out_nw, out_emb);

  // ---- attention pooling ----
  k_evidence<<<(Nc*128+255)/256, 256, 0, stream>>>(xc, non, nh_a);      // gelu(x)
  gemm(stream, SL1(seg_f32(qa, nullptr, 1024, 1024)), p_wq, bq, BSc, 1024, qsb, 0, 0);
  gemm(stream, SL1(seg_f16(nh_a, nullptr, 128, 128)), p_wk, bk, Nc, 128, recvb, 0, 0); // ks
  gemm(stream, SL1(seg_f16(nh_a, nullptr, 128, 128)), p_wv, bv, Nc, 128, vsb, 0, 0);   // vs
  k_attention<<<BSc, 256, 0, stream>>>(qsb, recvb, vsb, non, out_emb);
  k_copy_qa<<<(BSc*1024+255)/256, 256, 0, stream>>>(qa, out_emb);

  // ---- hid2out MLP on embeddings ----
  gemm(stream, SL1(seg_f32(out_emb, nullptr, EMBW, EMBW)), p_h1, hb1, BSc, 1280, sh_a, 1, 1);
  gemm(stream, SL1(seg_f16(sh_a, nullptr, 128, 128)), p_h2, hb2, BSc, 128, sh_b, 1, 1);
  k_gemv<<<1, 128, 0, stream>>>(sh_b, hW3, hb3, BSc, 128, 2, out_logits);
}
